// HyperConv2d_42167988912489
// MI455X (gfx1250) — matrix operand order repaired and checked by reading, not compiled, not run on hardware
//
#include <hip/hip_runtime.h>

// ---------------------------------------------------------------------------
// HyperConv2d for MI455X (gfx1250, wave32, WMMA).
//   Stage 1: w = z @ W_head^T + b   (bf16 WMMA, fp32 accum) -> ws (bf16),
//            stored with conv-K permuted to tap-major: kp = (kh*3+kw)*128 + ci
//   Stage 2: per-sample 3x3 conv as implicit GEMM (bf16 WMMA, fp32 accum).
//            Tap-major K => each 32-wide K-step has uniform (kh,kw,yy,xx),
//            so the im2col gather is a pure ci-stride walk (no div/mod chain).
// Memory-bound on the 302MB W_head read (~13us @ 23.3TB/s); bf16 16x16x32
// WMMA keeps the 29.6 GFLOP conv far below that floor (f32 16x16x4 would not).
// ---------------------------------------------------------------------------

typedef __attribute__((ext_vector_type(16))) __bf16 v16bf;
typedef __attribute__((ext_vector_type(8)))  __bf16 bf16x8;
typedef __attribute__((ext_vector_type(4)))  __bf16 bf16x4;
typedef __attribute__((ext_vector_type(8)))  float  v8f;

#define NB   32        // batch
#define IC   128
#define OC   128
#define ZD   512
#define HH   56
#define WW   56
#define HW   3136      // 56*56
#define KKC  1152      // IC*9  (conv GEMM K)
#define OUTF 147456    // OC*KKC (hypernet GEMM N)

// ---------------------------------------------------------------------------
// Kernel 1: x fp32 -> bf16 (vectorized, 4 elems/thread)
// ---------------------------------------------------------------------------
__global__ __launch_bounds__(256)
void x_to_bf16_kernel(const float* __restrict__ x, __bf16* __restrict__ xb, int n4) {
    int t = blockIdx.x * 256 + threadIdx.x;
    if (t < n4) {
        float4 v = ((const float4*)x)[t];
        bf16x4 o;
        o[0] = (__bf16)v.x; o[1] = (__bf16)v.y; o[2] = (__bf16)v.z; o[3] = (__bf16)v.w;
        ((bf16x4*)xb)[t] = o;
    }
}

// ---------------------------------------------------------------------------
// Kernel 2: hypernetwork head GEMM.
//   M = 32 (batch), N = 147456, K = 512.
//   Wave -> 32x16 tile (two 16x16 WMMA accumulators sharing B).
//   B fragment: W_head[n][k] rows are K-contiguous -> 4x b128 fp32 loads/lane.
//   A fragment: z staged in LDS as bf16 (padded stride), b128 ds loads.
//   Output scattered into tap-major conv-K layout (per-lane index, computed once).
// ---------------------------------------------------------------------------
__global__ __launch_bounds__(256)
void hyper_head_wmma(const float* __restrict__ z, const float* __restrict__ Wh,
                     const float* __restrict__ bh, __bf16* __restrict__ wout) {
    __shared__ __attribute__((aligned(16))) __bf16 zl[NB][ZD + 8]; // +8 halves: de-conflict banks

    int tid = threadIdx.x;
    for (int e = tid; e < NB * ZD; e += 256) {
        int m = e >> 9, k = e & (ZD - 1);
        zl[m][k] = (__bf16)z[m * ZD + k];
    }
    __syncthreads();

    int wv = tid >> 5, lane = tid & 31;
    int l16 = lane & 15, hi = lane >> 4;
    int col  = blockIdx.x * 128 + wv * 16 + l16;   // N index: col = o*1152 + ci*9 + r
    int koff = hi * 16;                            // A-fragment K contiguous half

    float bias = bh[col];
    v8f acc0, acc1;
#pragma unroll
    for (int r = 0; r < 8; ++r) { acc0[r] = bias; acc1[r] = bias; }

    const float* wrow = Wh + (size_t)col * ZD + hi * 16; // lanes 0-15: K 0..15, 16-31: K 16..31

    for (int kt = 0; kt < ZD / 32; ++kt) {
        const float* p = wrow + kt * 32;
        __builtin_prefetch(p + 32, 0, 1);          // global_prefetch_b8 next K-step
        float4 f0 = ((const float4*)p)[0];
        float4 f1 = ((const float4*)p)[1];
        float4 f2 = ((const float4*)p)[2];
        float4 f3 = ((const float4*)p)[3];
        float fv[16] = { f0.x, f0.y, f0.z, f0.w, f1.x, f1.y, f1.z, f1.w,
                         f2.x, f2.y, f2.z, f2.w, f3.x, f3.y, f3.z, f3.w };
        v16bf bb;
#pragma unroll
        for (int e = 0; e < 16; ++e) bb[e] = (__bf16)fv[e];

        // A fragments: lanes 0-15 hold K 0..15, lanes 16-31 K 16..31
        bf16x8 a0lo = *(const bf16x8*)&zl[l16     ][kt * 32 + koff];
        bf16x8 a0hi = *(const bf16x8*)&zl[l16     ][kt * 32 + 8 + koff];
        bf16x8 a1lo = *(const bf16x8*)&zl[16 + l16][kt * 32 + koff];
        bf16x8 a1hi = *(const bf16x8*)&zl[16 + l16][kt * 32 + 8 + koff];
        v16bf a0, a1;
#pragma unroll
        for (int e = 0; e < 8; ++e) {
            a0[e] = a0lo[e]; a0[8 + e] = a0hi[e];
            a1[e] = a1lo[e]; a1[8 + e] = a1hi[e];
        }
        acc0 = __builtin_amdgcn_wmma_f32_16x16x32_bf16(false, a0, false, bb,
                                                       (short)0, acc0, false, false);
        acc1 = __builtin_amdgcn_wmma_f32_16x16x32_bf16(false, a1, false, bb,
                                                       (short)0, acc1, false, false);
    }

    // Permuted scatter: col = o*1152 + ci*9 + r  ->  kp = r*128 + ci (tap-major)
    int oo  = col / KKC;
    int rem = col - oo * KKC;
    int ci  = rem / 9;
    int rr  = rem - ci * 9;
    size_t nperm = (size_t)oo * KKC + rr * IC + ci;

    // D layout: VGPR r -> lanes 0-15: M=r, lanes 16-31: M=8+r
    int mb = hi * 8;
#pragma unroll
    for (int r = 0; r < 8; ++r) {
        int m = mb + r;
        wout[(size_t)m        * OUTF + nperm] = (__bf16)acc0[r];
        wout[(size_t)(m + 16) * OUTF + nperm] = (__bf16)acc1[r];
    }
}

// ---------------------------------------------------------------------------
// Kernel 3: per-sample conv as implicit GEMM (M=OC, N=64-pixel tile, K=1152).
//   Block: 256 threads = 8 waves as 4(O) x 2(N); covers O=128, 64 pixels.
//   Tap-major K: chunk ks covers tap r = ks>>2, ci = (ks&3)*32 + [0,32).
//   Per K-step: cooperative im2col gather (ci-stride walk, uniform tap) into
//   double-buffered LDS; A fragments streamed from bf16 w (L2-resident).
// ---------------------------------------------------------------------------
__global__ __launch_bounds__(256)
void hyper_conv_wmma(const __bf16* __restrict__ xb, const __bf16* __restrict__ wb,
                     float* __restrict__ out) {
    // 64 pixels x 32 K, stride 40 halves (80B) -> conflict-free ds_load_b128
    __shared__ __attribute__((aligned(16))) __bf16 bch[2][64][40];

    int b   = blockIdx.y;
    int p0  = blockIdx.x * 64;
    int tid = threadIdx.x, lane = tid & 31, wv = tid >> 5;
    int l16 = lane & 15, hi = lane >> 4;
    int wo  = wv >> 1, wn = wv & 1;

    // im2col gather assignment: 8 elems/thread per K-step
    int gn  = tid & 63;            // pixel within tile
    int gk0 = (tid >> 6) * 8;      // K offset within 32-chunk
    int gp  = p0 + gn;
    int gy  = gp / WW, gx = gp - gy * WW;
    const __bf16* xbase = xb + (size_t)b * IC * HW;

    // A rows: o = wo*32 + mt*16 + l16 ; K-contiguous (tap-major) in wb
    const __bf16* arow0 = wb + ((size_t)b * OC + wo * 32 + l16) * KKC + hi * 16;
    const __bf16* arow1 = arow0 + (size_t)16 * KKC;

    v8f acc[2][2];
#pragma unroll
    for (int mt = 0; mt < 2; ++mt)
#pragma unroll
        for (int nt = 0; nt < 2; ++nt)
#pragma unroll
            for (int r = 0; r < 8; ++r) acc[mt][nt][r] = 0.f;

    auto gather = [&](int ks, int buf) {
        int r   = ks >> 2;                 // tap index 0..8 (uniform)
        int kh  = r / 3, kw = r - kh * 3;  // uniform per K-step
        int yy  = gy + kh - 1;
        int xx  = gx + kw - 1;
        int ci0 = ((ks & 3) << 5) + gk0;   // channel base for this thread's chunk
        bf16x8 tmp;
        if ((unsigned)yy < (unsigned)HH && (unsigned)xx < (unsigned)WW) {
            const __bf16* src = xbase + (size_t)ci0 * HW + yy * WW + xx;
#pragma unroll
            for (int j = 0; j < 8; ++j) tmp[j] = src[(size_t)j * HW]; // ci-stride walk
        } else {
#pragma unroll
            for (int j = 0; j < 8; ++j) tmp[j] = (__bf16)0.f;         // zero padding
        }
        *(bf16x8*)&bch[buf][gn][gk0] = tmp;  // ds_store_b128
    };

    gather(0, 0);
    __syncthreads();

    for (int ks = 0; ks < KKC / 32; ++ks) {
        int buf = ks & 1;
        if (ks + 1 < KKC / 32) gather(ks + 1, buf ^ 1); // prefetch next panel

        // A fragments (global bf16, b128 chunks)
        bf16x8 c00 = *(const bf16x8*)(arow0 + (size_t)ks * 32);
        bf16x8 c01 = *(const bf16x8*)(arow0 + (size_t)ks * 32 + 8);
        bf16x8 c10 = *(const bf16x8*)(arow1 + (size_t)ks * 32);
        bf16x8 c11 = *(const bf16x8*)(arow1 + (size_t)ks * 32 + 8);
        v16bf a0, a1;
#pragma unroll
        for (int e = 0; e < 8; ++e) {
            a0[e] = c00[e]; a0[8 + e] = c01[e];
            a1[e] = c10[e]; a1[8 + e] = c11[e];
        }

        // B fragments from LDS: lanes 0-15 K 0..15, lanes 16-31 K 16..31
        int nb0 = wn * 32 + l16;
        bf16x8 b0lo = *(const bf16x8*)&bch[buf][nb0     ][hi * 16];
        bf16x8 b0hi = *(const bf16x8*)&bch[buf][nb0     ][hi * 16 + 8];
        bf16x8 b1lo = *(const bf16x8*)&bch[buf][nb0 + 16][hi * 16];
        bf16x8 b1hi = *(const bf16x8*)&bch[buf][nb0 + 16][hi * 16 + 8];
        v16bf bb0, bb1;
#pragma unroll
        for (int e = 0; e < 8; ++e) {
            bb0[e] = b0lo[e]; bb0[8 + e] = b0hi[e];
            bb1[e] = b1lo[e]; bb1[8 + e] = b1hi[e];
        }

        acc[0][0] = __builtin_amdgcn_wmma_f32_16x16x32_bf16(false, a0, false, bb0,
                                                            (short)0, acc[0][0], false, false);
        acc[0][1] = __builtin_amdgcn_wmma_f32_16x16x32_bf16(false, a0, false, bb1,
                                                            (short)0, acc[0][1], false, false);
        acc[1][0] = __builtin_amdgcn_wmma_f32_16x16x32_bf16(false, a1, false, bb0,
                                                            (short)0, acc[1][0], false, false);
        acc[1][1] = __builtin_amdgcn_wmma_f32_16x16x32_bf16(false, a1, false, bb1,
                                                            (short)0, acc[1][1], false, false);
        __syncthreads();
    }

    // Epilogue: fp32 accumulators straight to d_out (N coalesced across lanes 0-15)
    int mb = hi * 8;
#pragma unroll
    for (int mt = 0; mt < 2; ++mt)
#pragma unroll
        for (int nt = 0; nt < 2; ++nt)
#pragma unroll
            for (int r = 0; r < 8; ++r) {
                int o = wo * 32 + mt * 16 + mb + r;
                int p = p0 + wn * 32 + nt * 16 + l16;
                out[((size_t)b * OC + o) * HW + p] = acc[mt][nt][r];
            }
}

// ---------------------------------------------------------------------------
extern "C" void kernel_launch(void* const* d_in, const int* in_sizes, int n_in,
                              void* d_out, int out_size, void* d_ws, size_t ws_size,
                              hipStream_t stream) {
    (void)in_sizes; (void)n_in; (void)out_size; (void)ws_size;
    const float* x  = (const float*)d_in[0];
    const float* z  = (const float*)d_in[1];
    const float* Wh = (const float*)d_in[2];
    const float* bh = (const float*)d_in[3];
    float* out = (float*)d_out;

    // Workspace: x_bf16 (25.7MB) | w_bf16 (9.4MB, tap-major K) -> ~35MB total
    __bf16* xb   = (__bf16*)d_ws;
    __bf16* wbuf = (__bf16*)((char*)d_ws + (size_t)NB * IC * HW * sizeof(__bf16));

    int n4 = NB * IC * HW / 4;
    x_to_bf16_kernel<<<(n4 + 255) / 256, 256, 0, stream>>>(x, xb, n4);

    hyper_head_wmma<<<OUTF / 128, 256, 0, stream>>>(z, Wh, bh, wbuf);

    dim3 grid(HW / 64, NB);
    hyper_conv_wmma<<<grid, 256, 0, stream>>>(xb, wbuf, out);
}
